// G_Attention_75265006895456
// MI455X (gfx1250) — matrix operand order repaired and checked by reading, not compiled, not run on hardware
//
#include <hip/hip_runtime.h>

// ---------------------------------------------------------------------------
// ViT attention block on gfx1250 (MI455X), f16 WMMA operands + f32 accumulate.
// B=64, N=197, DIM=768, H=12, DH=64.  M = B*N = 12608.
//
// Roofline: ~67 GFLOP vs ~100 MB unique data (L2-resident) => compute-bound;
// all matmuls ride v_wmma_f32_16x16x32_f16.  Operand range analysis (x~N(0,1),
// xavier-scaled weights) keeps every f16 operand orders of magnitude below
// overflow; accumulation and softmax stay fp32.
// ---------------------------------------------------------------------------

typedef __attribute__((ext_vector_type(16))) _Float16 v16h;
typedef __attribute__((ext_vector_type(8)))  float    v8f;

union Frag16 {
    v16h           v;
    unsigned short u[16];
    uint4          q[2];
};

__device__ __forceinline__ unsigned short f2h(float f) {
    union { _Float16 h; unsigned short u; } cv;
    cv.h = (_Float16)f;                       // v_cvt_f16_f32 (RNE)
    return cv.u;
}

__device__ __forceinline__ v8f zero8() {
    v8f z;
#pragma unroll
    for (int i = 0; i < 8; ++i) z[i] = 0.0f;
    return z;
}

__device__ __forceinline__ v8f wmma_f16(const Frag16& a, const Frag16& b, v8f c) {
    return __builtin_amdgcn_wmma_f32_16x16x32_f16(false, a.v, false, b.v,
                                                  (short)0, c, false, false);
}

// ---------------------------------------------------------------------------
// fp32 -> f16 conversion, 4 elements per thread.
// ---------------------------------------------------------------------------
__global__ __launch_bounds__(256) void k_f32_to_f16(const float* __restrict__ src,
                                                    unsigned short* __restrict__ dst,
                                                    int n4) {
    int i = blockIdx.x * 256 + threadIdx.x;
    if (i < n4) {
        float4 f = ((const float4*)src)[i];
        ushort4 o;
        o.x = f2h(f.x); o.y = f2h(f.y); o.z = f2h(f.z); o.w = f2h(f.w);
        ((ushort4*)dst)[i] = o;
    }
}

// ---------------------------------------------------------------------------
// Tiled f16 GEMM: C[M,N] = A[M,K] * B[K,N], f32 accumulate, double-buffered.
// 128x128 tile per 256-thread block (8 waves, each 32x64 = 2x4 WMMA tiles).
// MODE 0: epilogue scatters f16 into head-major q/k/v [B,H,197,64].
// MODE 1: epilogue adds bias, writes f32 to outf.
// ---------------------------------------------------------------------------
template<int MODE>
__global__ __launch_bounds__(256) void k_gemm_f16(
    const unsigned short* __restrict__ A, const unsigned short* __restrict__ Bm,
    int M, int N, int K,
    unsigned short* __restrict__ dq, unsigned short* __restrict__ dk,
    unsigned short* __restrict__ dv,
    const float* __restrict__ bias, float* __restrict__ outf)
{
    __shared__ unsigned short As[2][128 * 40];   // [m][k], row stride 40 halfs
    __shared__ unsigned short Bt[2][128 * 40];   // [n][k], row stride 40 halfs

    const int tid  = threadIdx.x;
    const int wave = tid >> 5;
    const int lane = tid & 31;
    const int ln   = lane & 15;
    const int hf   = lane >> 4;
    const int wr   = wave & 3;                   // 4 wave-rows of 32
    const int wc   = wave >> 2;                  // 2 wave-cols of 64
    const int m0   = blockIdx.y * 128;
    const int n0   = blockIdx.x * 128;
    const int nk   = K >> 5;

    v8f acc[2][4];
#pragma unroll
    for (int rt = 0; rt < 2; ++rt)
#pragma unroll
        for (int ct = 0; ct < 4; ++ct) acc[rt][ct] = zero8();

    // --- prologue: stage K-step 0 into buffer 0 ---
#pragma unroll
    for (int it = 0; it < 2; ++it) {
        int c = tid + it * 256;
        int row = c >> 2, ch = c & 3;
        int gm  = m0 + row;
        uint4 val = make_uint4(0u, 0u, 0u, 0u);
        if (gm < M) val = *(const uint4*)(A + (size_t)gm * K + ch * 8);
        *(uint4*)(&As[0][row * 40 + ch * 8]) = val;
    }
#pragma unroll
    for (int it = 0; it < 2; ++it) {
        int c = tid + it * 256;
        int kk = c >> 4, nch = c & 15;
        uint4 val = *(const uint4*)(Bm + (size_t)kk * N + n0 + nch * 8);
        const unsigned short* pv = (const unsigned short*)&val;
#pragma unroll
        for (int i = 0; i < 8; ++i) Bt[0][(nch * 8 + i) * 40 + kk] = pv[i];
    }
    __syncthreads();

    for (int ks = 0; ks < nk; ++ks) {
        const int  cur  = ks & 1;
        const int  nxt  = cur ^ 1;
        const bool more = (ks + 1 < nk);

        // --- prefetch next K-step (global -> regs), overlaps with WMMAs ---
        uint4 pa[2], pb[2];
        if (more) {
            const int k0 = (ks + 1) << 5;
#pragma unroll
            for (int it = 0; it < 2; ++it) {
                int c = tid + it * 256;
                int row = c >> 2, ch = c & 3;
                int gm  = m0 + row;
                pa[it] = make_uint4(0u, 0u, 0u, 0u);
                if (gm < M) pa[it] = *(const uint4*)(A + (size_t)gm * K + k0 + ch * 8);
            }
#pragma unroll
            for (int it = 0; it < 2; ++it) {
                int c = tid + it * 256;
                int kk = c >> 4, nch = c & 15;
                pb[it] = *(const uint4*)(Bm + (size_t)(k0 + kk) * N + n0 + nch * 8);
            }
        }

        // --- fragments (A layout: lane half holds contiguous K = hf*16 + e for row m=ln;
        //     B layout: lane half holds contiguous K = hf*16 + e for column n=ln) ---
        Frag16 af[2], bf[4];
#pragma unroll
        for (int rt = 0; rt < 2; ++rt) {
            int ml = wr * 32 + rt * 16 + ln;
            af[rt].q[0] = *(const uint4*)(&As[cur][ml * 40 + hf * 16]);
            af[rt].q[1] = *(const uint4*)(&As[cur][ml * 40 + hf * 16 + 8]);
        }
#pragma unroll
        for (int ct = 0; ct < 4; ++ct) {
            int nl = wc * 64 + ct * 16 + ln;
            bf[ct].q[0] = *(const uint4*)(&Bt[cur][nl * 40 + hf * 16]);
            bf[ct].q[1] = *(const uint4*)(&Bt[cur][nl * 40 + hf * 16 + 8]);
        }
#pragma unroll
        for (int rt = 0; rt < 2; ++rt)
#pragma unroll
            for (int ct = 0; ct < 4; ++ct)
                acc[rt][ct] = wmma_f16(af[rt], bf[ct], acc[rt][ct]);

        // --- drain prefetch into the other buffer (safe: peers read buf[cur]) ---
        if (more) {
#pragma unroll
            for (int it = 0; it < 2; ++it) {
                int c = tid + it * 256;
                int row = c >> 2, ch = c & 3;
                *(uint4*)(&As[nxt][row * 40 + ch * 8]) = pa[it];
            }
#pragma unroll
            for (int it = 0; it < 2; ++it) {
                int c = tid + it * 256;
                int kk = c >> 4, nch = c & 15;
                const unsigned short* pv = (const unsigned short*)&pb[it];
#pragma unroll
                for (int i = 0; i < 8; ++i) Bt[nxt][(nch * 8 + i) * 40 + kk] = pv[i];
            }
        }
        __syncthreads();
    }

    // --- epilogue (C layout: row = r + 8*hf, col = ln within each 16x16 tile) ---
#pragma unroll
    for (int rt = 0; rt < 2; ++rt) {
#pragma unroll
        for (int ct = 0; ct < 4; ++ct) {
            const int gn = n0 + wc * 64 + ct * 16 + ln;
            if (MODE == 0) {
                const int which = gn / 768;
                const int rem   = gn - which * 768;
                const int hh    = rem >> 6;
                const int dd    = rem & 63;
                unsigned short* dst = (which == 0) ? dq : (which == 1) ? dk : dv;
#pragma unroll
                for (int r = 0; r < 8; ++r) {
                    const int gm = m0 + wr * 32 + rt * 16 + r + 8 * hf;
                    if (gm < M) {
                        const int bidx = gm / 197;
                        const int seq  = gm - bidx * 197;
                        dst[(((size_t)bidx * 12 + hh) * 197 + seq) * 64 + dd] =
                            f2h(acc[rt][ct][r]);
                    }
                }
            } else {
                const float bv = bias[gn];
#pragma unroll
                for (int r = 0; r < 8; ++r) {
                    const int gm = m0 + wr * 32 + rt * 16 + r + 8 * hf;
                    if (gm < M) outf[(size_t)gm * N + gn] = acc[rt][ct][r] + bv;
                }
            }
        }
    }
}

// ---------------------------------------------------------------------------
// Per-(b,h) masked-softmax attention.  Block = (rb, h, b): 64 Q rows, 4 waves,
// each wave owns one 16-row strip.
//   Ks [row][72]  : K rows 0..207 (zero >=197); aliased as probs after dots.
//   Vt [dh][232]  : V transposed, rows 0..223 (zero >=197) so PV B-fragments
//                   are contiguous ds_load_b128 pairs.
//   probs [16][232] per wave, cols 0..223 (208..223 zeroed) so PV A-fragments
//                   are contiguous ds_load_b128 pairs.
// ---------------------------------------------------------------------------
__global__ __launch_bounds__(128) void k_attn(
    const unsigned short* __restrict__ Qg, const unsigned short* __restrict__ Kg,
    const unsigned short* __restrict__ Vg, const float* __restrict__ scale,
    unsigned short* __restrict__ outc)
{
    __shared__ unsigned short Ks[208 * 72];    // 29952 B; later probs (4*16*232)
    __shared__ unsigned short Vt[64 * 232];    // 29696 B

    const int tid  = threadIdx.x;
    const int wave = tid >> 5;
    const int lane = tid & 31;
    const int ln   = lane & 15;
    const int hf   = lane >> 4;
    const int rb   = blockIdx.x;
    const int h    = blockIdx.y;
    const int b    = blockIdx.z;
    const size_t base = ((size_t)(b * 12 + h)) * 197 * 64;

    // stage K rows (zero-padded to 208), coalesced 16B chunks
    for (int c = tid; c < 208 * 8; c += 128) {
        int row = c >> 3, c8 = c & 7;
        uint4 kv = make_uint4(0u, 0u, 0u, 0u);
        if (row < 197) kv = *(const uint4*)(Kg + base + (size_t)row * 64 + c8 * 8);
        *(uint4*)(&Ks[row * 72 + c8 * 8]) = kv;
    }
    // stage V transposed: cover rows 0..223 (zero >=197) for all 64 dh cols
    for (int c = tid; c < 224 * 8; c += 128) {
        int row = c >> 3, c8 = c & 7;
        uint4 vv = make_uint4(0u, 0u, 0u, 0u);
        if (row < 197) vv = *(const uint4*)(Vg + base + (size_t)row * 64 + c8 * 8);
        const unsigned short* pv = (const unsigned short*)&vv;
#pragma unroll
        for (int i = 0; i < 8; ++i) Vt[(c8 * 8 + i) * 232 + row] = pv[i];
    }

    // Q fragments straight from global (two 16B chunks per K-step, A layout)
    const int strip = rb * 64 + wave * 16;
    int qrow = strip + ln; if (qrow > 196) qrow = 196;
    Frag16 aq[2];
#pragma unroll
    for (int kt = 0; kt < 2; ++kt) {
        aq[kt].q[0] = *(const uint4*)(Qg + base + (size_t)qrow * 64 + kt * 32 + hf * 16);
        aq[kt].q[1] = *(const uint4*)(Qg + base + (size_t)qrow * 64 + kt * 32 + hf * 16 + 8);
    }
    __syncthreads();

    // dots = Q @ K^T : 13 tiles of 16 cols, K-dim 64 in two 32-steps
    v8f acc[13];
#pragma unroll
    for (int jt = 0; jt < 13; ++jt) {
        v8f c = zero8();
#pragma unroll
        for (int kt = 0; kt < 2; ++kt) {
            Frag16 bk;   // B[k][n] = K[jt*16+n][kt*32+k]; contiguous 16 halfs/lane
            bk.q[0] = *(const uint4*)(&Ks[(jt * 16 + ln) * 72 + kt * 32 + hf * 16]);
            bk.q[1] = *(const uint4*)(&Ks[(jt * 16 + ln) * 72 + kt * 32 + hf * 16 + 8]);
            c = wmma_f16(aq[kt], bk, c);
        }
        acc[jt] = c;
    }
    __syncthreads();   // all waves finished reading Ks -> safe to alias as probs

    // softmax (scale, diagonal mask, length mask); probs -> aliased LDS (f16)
    unsigned short* probs = Ks + wave * 16 * 232;   // wave-private [16][232]
    const float sc = scale[h];
#pragma unroll
    for (int r = 0; r < 8; ++r) {
        const int lrow = r + 8 * hf;
        const int grow = strip + lrow;
        float m = -3.0e38f;
#pragma unroll
        for (int jt = 0; jt < 13; ++jt) {
            float vv = acc[jt][r] * sc;
            const int gcol = jt * 16 + ln;
            if (gcol >= 197 || gcol == grow) vv = -3.0e38f;
            acc[jt][r] = vv;
            m = fmaxf(m, vv);
        }
#pragma unroll
        for (int off = 1; off < 16; off <<= 1) m = fmaxf(m, __shfl_xor(m, off, 32));
        float s = 0.0f;
#pragma unroll
        for (int jt = 0; jt < 13; ++jt) {
            float p = __expf(acc[jt][r] - m);
            acc[jt][r] = p;
            s += p;
        }
#pragma unroll
        for (int off = 1; off < 16; off <<= 1) s += __shfl_xor(s, off, 32);
        const float inv = 1.0f / s;
#pragma unroll
        for (int jt = 0; jt < 13; ++jt)
            probs[lrow * 232 + jt * 16 + ln] = f2h(acc[jt][r] * inv);
        probs[lrow * 232 + 208 + ln] = 0;            // zero pad cols 208..223
    }

    // out = P @ V : 7 K-tiles of 32, 4 DH col-tiles; all-b128 LDS fragments
    v8f oacc[4];
#pragma unroll
    for (int ct = 0; ct < 4; ++ct) oacc[ct] = zero8();
#pragma unroll
    for (int kt = 0; kt < 7; ++kt) {
        Frag16 ap;   // A layout over probs row ln
        ap.q[0] = *(const uint4*)(probs + ln * 232 + kt * 32 + hf * 16);
        ap.q[1] = *(const uint4*)(probs + ln * 232 + kt * 32 + hf * 16 + 8);
#pragma unroll
        for (int ct = 0; ct < 4; ++ct) {
            Frag16 bv;   // B[k][n] = V[kt*32+k][ct*16+n] = Vt[n][row], contiguous
            bv.q[0] = *(const uint4*)(&Vt[(ct * 16 + ln) * 232 + kt * 32 + hf * 16]);
            bv.q[1] = *(const uint4*)(&Vt[(ct * 16 + ln) * 232 + kt * 32 + hf * 16 + 8]);
            oacc[ct] = wmma_f16(ap, bv, oacc[ct]);
        }
    }
#pragma unroll
    for (int ct = 0; ct < 4; ++ct) {
#pragma unroll
        for (int r = 0; r < 8; ++r) {
            const int grow = strip + r + 8 * hf;
            if (grow < 197)
                outc[((size_t)(b * 197 + grow)) * 768 + h * 64 + ct * 16 + ln] =
                    f2h(oacc[ct][r]);
        }
    }
}

// ---------------------------------------------------------------------------
extern "C" void kernel_launch(void* const* d_in, const int* in_sizes, int n_in,
                              void* d_out, int out_size, void* d_ws, size_t ws_size,
                              hipStream_t stream) {
    const float* x     = (const float*)d_in[0];
    const float* Wqkv  = (const float*)d_in[1];
    const float* scale = (const float*)d_in[2];
    const float* Wout  = (const float*)d_in[3];
    const float* bout  = (const float*)d_in[4];
    float* out = (float*)d_out;

    const int    M   = 64 * 197;              // 12608
    const size_t nX  = (size_t)M * 768;       // 9,682,944
    const size_t nW1 = (size_t)768 * 2304;    // 1,769,472
    const size_t nW2 = (size_t)768 * 768;     //   589,824

    // workspace layout (f16 halves); attention output aliases xb, which is
    // dead after GEMM1 (stream-ordered, so this is safe and deterministic).
    unsigned short* ws  = (unsigned short*)d_ws;
    unsigned short* xb  = ws;  ws += nX;
    unsigned short* w1b = ws;  ws += nW1;
    unsigned short* w2b = ws;  ws += nW2;
    unsigned short* q   = ws;  ws += nX;
    unsigned short* k   = ws;  ws += nX;
    unsigned short* v   = ws;  ws += nX;
    unsigned short* att = xb;                 // alias

    k_f32_to_f16<<<(int)((nX / 4 + 255) / 256), 256, 0, stream>>>(x, xb, (int)(nX / 4));
    k_f32_to_f16<<<(int)((nW1 / 4 + 255) / 256), 256, 0, stream>>>(Wqkv, w1b, (int)(nW1 / 4));
    k_f32_to_f16<<<(int)((nW2 / 4 + 255) / 256), 256, 0, stream>>>(Wout, w2b, (int)(nW2 / 4));

    // qkv = x @ W_qkv, scattered into head-major q/k/v
    k_gemm_f16<0><<<dim3(2304 / 128, (M + 127) / 128), 256, 0, stream>>>(
        xb, w1b, M, 2304, 768, q, k, v, nullptr, nullptr);

    // per-head masked softmax attention
    k_attn<<<dim3(4, 12, 64), 128, 0, stream>>>(q, k, v, scale, att);

    // out = att @ W_out + b_out (f32)
    k_gemm_f16<1><<<dim3(768 / 128, (M + 127) / 128), 256, 0, stream>>>(
        att, w2b, M, 768, 768, nullptr, nullptr, nullptr, bout, out);
}
